// Masker_23888608101162
// MI455X (gfx1250) — matrix operand order repaired and checked by reading, not compiled, not run on hardware
//
#include <hip/hip_runtime.h>
#include <cstdint>
#include <cstddef>

#define IN_DIM       2048
#define MIDDLE       8192
#define NUM_CLASSES  2048
#define BATCH        32
#define KSTEPS       1024
#define BN_EPS       1e-5f

typedef __attribute__((ext_vector_type(16))) _Float16 v16h;
typedef __attribute__((ext_vector_type(8)))  float    v8f;
typedef __attribute__((ext_vector_type(4)))  uint32_t u32x4;
typedef __attribute__((ext_vector_type(8)))  int      i32x8;
typedef __attribute__((ext_vector_type(4)))  int      i32x4;

#define NTILE 128            // output columns per workgroup
#define KC    32             // K-chunk per TDM transfer (one WMMA K-step)
#define WROW  (NTILE + 1)    // padded LDS row stride (floats) for W tile
#define AROW  (KC + 1)       // padded LDS row stride (floats) for A tile
#define CTS   (NTILE + 4)    // epilogue tile stride

#define WT_FLOATS (2 * KC * WROW)      // 8256 floats (~33 KB), double-buffered W
#define AS_FLOATS (2 * BATCH * AROW)   // 2112 floats (~8.4 KB), double-buffered A

#if __has_builtin(__builtin_amdgcn_tensor_load_to_lds) && \
    __has_builtin(__builtin_amdgcn_s_wait_tensorcnt)
#define USE_TDM 1
#else
#define USE_TDM 0
#endif

#if USE_TDM
// Issue one 2D TDM tile load: tile_d1 rows x tile_d0 elems (fp32) from a
// row-major tensor with row stride stride_d0, into LDS at lds_byte_addr with
// +1-DWORD padding per tile row (pad_interval_code: 2^(c+1) DWORDs).
__device__ __forceinline__ void tdm_load_2d(
    uint32_t lds_byte_addr, const float* gptr,
    uint32_t tile_d0, uint32_t tile_d1,
    uint32_t tensor_d0, uint32_t tensor_d1,
    uint32_t stride_d0, uint32_t pad_interval_code)
{
  const uint64_t ga = (uint64_t)(uintptr_t)gptr;
  u32x4 g0;
  g0[0] = 1u;                                 // count=1 (valid user descriptor)
  g0[1] = lds_byte_addr;                      // D#.lds_addr
  g0[2] = (uint32_t)ga;                       // D#.global_addr[31:0]
  g0[3] = (uint32_t)(ga >> 32) | (2u << 30);  // global_addr[56:32] | type=2
  i32x8 g1;
  g1[0] = (int)((2u << 16)                    // data_size = 4 bytes
              | (1u << 20)                    // pad_enable
              | (pad_interval_code << 22));   // pad every 2^(c+1) DWORDs, +1 DWORD
  g1[1] = (int)((tensor_d0 & 0xFFFFu) << 16);                       // dim0[15:0]
  g1[2] = (int)((tensor_d0 >> 16) | ((tensor_d1 & 0xFFFFu) << 16)); // dim0 hi | dim1 lo
  g1[3] = (int)((tensor_d1 >> 16) | (tile_d0 << 16));               // dim1 hi | tile0
  g1[4] = (int)tile_d1;                       // tile_dim1 | tile_dim2=0
  g1[5] = (int)stride_d0;                     // tensor_dim0_stride[31:0]
  g1[6] = 0;                                  // stride hi | dim1_stride lo
  g1[7] = 0;
  i32x4 z4 = {0, 0, 0, 0};
#if __clang_major__ >= 23
  i32x8 z8 = {0, 0, 0, 0, 0, 0, 0, 0};
  __builtin_amdgcn_tensor_load_to_lds(g0, g1, z4, z4, z8, 0);
#else
  __builtin_amdgcn_tensor_load_to_lds(g0, g1, z4, z4, 0);
#endif
}
#endif

// Fused  out = [relu(bn(] A@W + bias [))]  with A=[32,K], W=[K,N].
// One workgroup = full batch (M=32) x 128 cols; 8 waves, each owning a
// 16-col slice with two f16 WMMA accumulators. A and W chunks are streamed
// HBM->LDS by the Tensor Data Mover, double-buffered, sync'd by TENSORcnt.
__global__ __launch_bounds__(256) void gemm_bias_bn_relu(
    const float* __restrict__ A,
    const float* __restrict__ W,
    const float* __restrict__ bias,
    const float* __restrict__ gamma,
    const float* __restrict__ beta,
    float* __restrict__ out,
    int K, int N, int fuse_bn)
{
  __shared__ __align__(16) float smem[WT_FLOATS + AS_FLOATS];
  float* Wt  = smem;                 // [2][KC][WROW]
  float* Asm = smem + WT_FLOATS;     // [2][BATCH][AROW]
  float* Ct  = smem;                 // aliased epilogue tile [BATCH][CTS]

  const int t    = threadIdx.x;
  const int lane = t & 31;
  const int wave = t >> 5;
  const int half = lane >> 4;
  const int l16  = lane & 15;
  const int colbase = blockIdx.x * NTILE;
  const int wcol    = wave * 16;
  const int nchunk  = K / KC;

  v8f acc0 = {};   // rows 0..15
  v8f acc1 = {};   // rows 16..31

#if USE_TDM
  const uint32_t wt_base = (uint32_t)(uintptr_t)Wt;   // LDS byte offset
  const uint32_t as_base = (uint32_t)(uintptr_t)Asm;
  if (t < 32) {   // wave 0 drives the TDM pipeline
    tdm_load_2d(as_base, A, KC, BATCH, (uint32_t)K, BATCH, (uint32_t)K, 4);
    tdm_load_2d(wt_base, W + colbase, NTILE, KC, (uint32_t)N, (uint32_t)K,
                (uint32_t)N, 6);
  }
#endif

  for (int ch = 0; ch < nchunk; ++ch) {
    const int cur = ch & 1;
    float* Wc = Wt  + cur * (KC * WROW);
    float* Ac = Asm + cur * (BATCH * AROW);

#if USE_TDM
    if (t < 32) {
      if (ch + 1 < nchunk) {
        const int nxt = 1 - cur;
        tdm_load_2d(as_base + (uint32_t)nxt * (BATCH * AROW * 4),
                    A + (size_t)(ch + 1) * KC,
                    KC, BATCH, (uint32_t)K, BATCH, (uint32_t)K, 4);
        tdm_load_2d(wt_base + (uint32_t)nxt * (KC * WROW * 4),
                    W + (size_t)(ch + 1) * KC * N + colbase,
                    NTILE, KC, (uint32_t)N, (uint32_t)K, (uint32_t)N, 6);
        __builtin_amdgcn_s_wait_tensorcnt(2);   // 2 newest in flight, chunk ch done
      } else {
        __builtin_amdgcn_s_wait_tensorcnt(0);
      }
    }
#else
    for (int q = t; q < BATCH * KC; q += 256) {
      int r = q / KC, c = q % KC;
      Ac[r * AROW + c] = A[(size_t)r * K + (size_t)ch * KC + c];
    }
    for (int q = t; q < KC * NTILE; q += 256) {
      int r = q / NTILE, c = q % NTILE;
      Wc[r * WROW + c] = W[((size_t)ch * KC + r) * (size_t)N + colbase + c];
    }
#endif
    __syncthreads();

    // B fragment: 32x16 f16; e<8 -> K=half*8+e, e>=8 -> K=16+half*8+(e-8)
    v16h b;
    #pragma unroll
    for (int e = 0; e < 16; ++e)
      b[e] = (_Float16)Wc[((e & 8) * 2 + half * 8 + (e & 7)) * WROW + wcol + l16];

    // A fragments: 16x32 f16; e<8 -> K=koff+e, e>=8 -> K=16+koff+(e-8)
    const int koff = half * 8;
    v16h a0, a1;
    #pragma unroll
    for (int e = 0; e < 8; ++e) {
      a0[e]     = (_Float16)Ac[l16 * AROW + koff + e];
      a0[e + 8] = (_Float16)Ac[l16 * AROW + koff + 16 + e];
      a1[e]     = (_Float16)Ac[(l16 + 16) * AROW + koff + e];
      a1[e + 8] = (_Float16)Ac[(l16 + 16) * AROW + koff + 16 + e];
    }

    acc0 = __builtin_amdgcn_wmma_f32_16x16x32_f16(
        false, a0, false, b, (short)0, acc0, false, false);
    acc1 = __builtin_amdgcn_wmma_f32_16x16x32_f16(
        false, a1, false, b, (short)0, acc1, false, false);

    __syncthreads();
  }

  // ---- epilogue: scatter accumulators to (aliased) LDS tile [32][NTILE] ----
  // D layout: VGPR r, lane l -> row = r + 8*(l>>4), col = l&15
  #pragma unroll
  for (int r = 0; r < 8; ++r) {
    Ct[(r + half * 8) * CTS + wcol + l16]        = acc0[r];
    Ct[(r + half * 8 + 16) * CTS + wcol + l16]   = acc1[r];
  }
  __syncthreads();

  if (t < NTILE) {
    const int gc = colbase + t;
    const float bv = bias[gc];
    if (fuse_bn) {
      float m = 0.0f, m2 = 0.0f;
      #pragma unroll
      for (int r = 0; r < BATCH; ++r) {
        float x = Ct[r * CTS + t] + bv;
        m += x; m2 += x * x;
      }
      m  *= (1.0f / BATCH);
      m2 *= (1.0f / BATCH);
      float var = m2 - m * m;                  // biased variance (torch BN)
      float sc  = rsqrtf(var + BN_EPS) * gamma[gc];
      float sh  = beta[gc] - m * sc;
      #pragma unroll
      for (int r = 0; r < BATCH; ++r) {
        float y = (Ct[r * CTS + t] + bv) * sc + sh;
        out[(size_t)r * N + gc] = fmaxf(y, 0.0f);
      }
    } else {
      #pragma unroll
      for (int r = 0; r < BATCH; ++r)
        out[(size_t)r * N + gc] = Ct[r * CTS + t] + bv;
    }
  }
}

// ---------------- Gumbel top-K masking ----------------
__device__ __forceinline__ uint32_t pcg_hash(uint32_t x) {
  x ^= x >> 16; x *= 0x7feb352dU;
  x ^= x >> 15; x *= 0x846ca68bU;
  x ^= x >> 16;
  return x;
}

#define GPER (NUM_CLASSES / 256)   // 8 columns per thread

__global__ __launch_bounds__(256) void gumbel_topk_mask(
    const float* __restrict__ mask, float* __restrict__ z_out)
{
  const int row  = blockIdx.x;
  const int t    = threadIdx.x;
  const int lane = t & 31;
  const int wave = t >> 5;

  float rm[GPER], z[GPER];
  #pragma unroll
  for (int j = 0; j < GPER; ++j) {
    rm[j] = mask[(size_t)row * NUM_CLASSES + t + 256 * j];
    z[j]  = 0.0f;
  }

  __shared__ float redv[8];
  __shared__ int   redi[8];
  __shared__ float reds[8];
  __shared__ float s_max, s_sum;
  __shared__ int   s_arg;

  for (int i = 0; i < KSTEPS; ++i) {
    float v[GPER];
    float vmax = -__builtin_inff();
    int   varg = 0;
    #pragma unroll
    for (int j = 0; j < GPER; ++j) {
      int c = t + 256 * j;
      uint32_t h = pcg_hash((uint32_t)i * 0x9E3779B1u ^
                            (uint32_t)row * 0x85EBCA77u ^
                            (uint32_t)c * 0xC2B2AE3Du);
      float u = ((float)(h >> 8) + 0.5f) * (1.0f / 16777216.0f);
      float g = -__logf(-__logf(u));
      v[j] = (rm[j] + g) * 2.0f;               // 1/tau
      if (v[j] > vmax) { vmax = v[j]; varg = c; }
    }
    #pragma unroll
    for (int o = 16; o > 0; o >>= 1) {
      float ov = __shfl_xor(vmax, o, 32);
      int   oi = __shfl_xor(varg, o, 32);
      if (ov > vmax || (ov == vmax && oi < varg)) { vmax = ov; varg = oi; }
    }
    if (lane == 0) { redv[wave] = vmax; redi[wave] = varg; }
    __syncthreads();
    if (t == 0) {
      float bm = redv[0]; int ba = redi[0];
      #pragma unroll
      for (int w = 1; w < 8; ++w)
        if (redv[w] > bm || (redv[w] == bm && redi[w] < ba)) { bm = redv[w]; ba = redi[w]; }
      s_max = bm; s_arg = ba;
    }
    __syncthreads();
    const float mrow = s_max;
    const int   amax = s_arg;

    float e[GPER], lsum = 0.0f;
    #pragma unroll
    for (int j = 0; j < GPER; ++j) { e[j] = __expf(v[j] - mrow); lsum += e[j]; }
    #pragma unroll
    for (int o = 16; o > 0; o >>= 1) lsum += __shfl_xor(lsum, o, 32);
    if (lane == 0) reds[wave] = lsum;
    __syncthreads();
    if (t == 0) {
      float s = 0.0f;
      #pragma unroll
      for (int w = 0; w < 8; ++w) s += reds[w];
      s_sum = s;
    }
    __syncthreads();
    const float inv = 1.0f / s_sum;
    #pragma unroll
    for (int j = 0; j < GPER; ++j) {
      float s = e[j] * inv;
      z[j] = fmaxf(z[j], s);
      if (t + 256 * j == amax) rm[j] = -__builtin_inff();
    }
    __syncthreads();
  }

  #pragma unroll
  for (int j = 0; j < GPER; ++j)
    z_out[(size_t)row * NUM_CLASSES + t + 256 * j] = z[j];
}

extern "C" void kernel_launch(void* const* d_in, const int* in_sizes, int n_in,
                              void* d_out, int out_size, void* d_ws, size_t ws_size,
                              hipStream_t stream) {
  (void)in_sizes; (void)n_in; (void)out_size; (void)ws_size;
  const float* f   = (const float*)d_in[0];
  const float* W1  = (const float*)d_in[1];
  const float* b1  = (const float*)d_in[2];
  const float* g1  = (const float*)d_in[3];
  const float* bt1 = (const float*)d_in[4];
  const float* W2  = (const float*)d_in[5];
  const float* b2  = (const float*)d_in[6];
  const float* g2  = (const float*)d_in[7];
  const float* bt2 = (const float*)d_in[8];
  const float* W3  = (const float*)d_in[9];
  const float* b3  = (const float*)d_in[10];

  float* ws = (float*)d_ws;
  float* h1 = ws;                                  // [32, 8192]
  float* h2 = ws + (size_t)BATCH * MIDDLE;         // [32, 8192]
  float* mk = ws + (size_t)2 * BATCH * MIDDLE;     // [32, 2048]

  dim3 blk(256);
  gemm_bias_bn_relu<<<MIDDLE / NTILE, blk, 0, stream>>>(
      f, W1, b1, g1, bt1, h1, IN_DIM, MIDDLE, 1);
  gemm_bias_bn_relu<<<MIDDLE / NTILE, blk, 0, stream>>>(
      h1, W2, b2, g2, bt2, h2, MIDDLE, MIDDLE, 1);
  gemm_bias_bn_relu<<<NUM_CLASSES / NTILE, blk, 0, stream>>>(
      h2, W3, b3, nullptr, nullptr, mk, MIDDLE, NUM_CLASSES, 0);
  gumbel_topk_mask<<<BATCH, blk, 0, stream>>>(mk, (float*)d_out);
}
